// TorchVisionAttention_89713276879660
// MI455X (gfx1250) — matrix operand order repaired and checked by reading, not compiled, not run on hardware
//
#include <hip/hip_runtime.h>
#include <stdint.h>

#ifndef __has_builtin
#define __has_builtin(x) 0
#endif

#if defined(__gfx1250__) && __has_builtin(__builtin_amdgcn_global_load_async_to_lds_b128)
#define USE_ASYNC_LDS 1
#else
#define USE_ASYNC_LDS 0
#endif

typedef __bf16 bf16_t;
typedef __attribute__((ext_vector_type(16))) __bf16 bf16x16;
typedef __attribute__((ext_vector_type(8)))  __bf16 bf16x8;
typedef __attribute__((ext_vector_type(4)))  __bf16 bf16x4;
typedef __attribute__((ext_vector_type(8)))  float  f32x8;
typedef __attribute__((ext_vector_type(4)))  float  f32x4;
typedef __attribute__((ext_vector_type(4)))  int    i32x4;

union Frag {
  bf16x16  v;
  bf16x8   h[2];
  uint32_t w[8];
};

constexpr int Bn = 8, Sn = 1024, En = 1280, Hn = 16, Dn = 80;
constexpr int Mn = Bn * Sn;  // 8192 rows

__device__ __forceinline__ f32x8 zf8() {
  f32x8 z = {0.f, 0.f, 0.f, 0.f, 0.f, 0.f, 0.f, 0.f};
  return z;
}

// 16-byte global -> LDS copy via gfx1250 async-to-LDS (ASYNCcnt tracked).
__device__ __forceinline__ void g2l_16B(const uint16_t* g, uint16_t* l) {
#if USE_ASYNC_LDS
  __builtin_amdgcn_global_load_async_to_lds_b128((i32x4*)g, (i32x4*)l, 0, 0);
#else
  *(f32x4*)l = *(const f32x4*)g;
#endif
}

__device__ __forceinline__ void wait_g2l() {
#if USE_ASYNC_LDS
  asm volatile("s_wait_asynccnt 0" ::: "memory");
#endif
}

// ---------------------------------------------------------------------------
// DPP-based 16-lane row reductions (replace ds_bpermute shuffles with VALU).
// Stages: quad swap (xor1), quad rotate-2 (xor2), row_half_mirror, row_mirror.
// ---------------------------------------------------------------------------
template<int CTRL>
__device__ __forceinline__ float dpp_perm(float x) {
  return __builtin_bit_cast(float, __builtin_amdgcn_update_dpp(
      0, __builtin_bit_cast(int, x), CTRL, 0xf, 0xf, true));
}
__device__ __forceinline__ float rowmax16(float x) {
  x = fmaxf(x, dpp_perm<0xB1>(x));    // quad_perm [1,0,3,2]
  x = fmaxf(x, dpp_perm<0x4E>(x));    // quad_perm [2,3,0,1]
  x = fmaxf(x, dpp_perm<0x141>(x));   // row_half_mirror
  x = fmaxf(x, dpp_perm<0x140>(x));   // row_mirror
  return x;
}
__device__ __forceinline__ float rowsum16(float x) {
  x += dpp_perm<0xB1>(x);
  x += dpp_perm<0x4E>(x);
  x += dpp_perm<0x141>(x);
  x += dpp_perm<0x140>(x);
  return x;
}

// ---------------------------------------------------------------------------
// fp32 -> bf16 conversion (x and the four weight matrices), 4 elems/thread
// ---------------------------------------------------------------------------
__global__ void cvt_f32_to_bf16(const float* __restrict__ src,
                                uint16_t* __restrict__ dst, int n4) {
  int i = blockIdx.x * blockDim.x + threadIdx.x;
  if (i >= n4) return;
  f32x4 f = *(const f32x4*)(src + 4l * i);
  bf16x4 o;
  o[0] = (bf16_t)f[0]; o[1] = (bf16_t)f[1];
  o[2] = (bf16_t)f[2]; o[3] = (bf16_t)f[3];
  *(bf16x4*)(dst + 4l * i) = o;
}

// ---------------------------------------------------------------------------
// Y = A (Mn x En, bf16 rows) * W^T.  BM=128, BN=128, BK=32; 8 waves (4m x 2n),
// each wave a 32x64 tile = 2x4 WMMA accumulators (8 WMMA / k-step / wave).
// MODE 0: bf16 row-major [Mn][En]       (Q, K)
// MODE 1: bf16 transposed Vt[b][f][s]   (V; PV B-frags become contiguous)
// MODE 2: f32 row-major [Mn][En]        (final output)
// ---------------------------------------------------------------------------
template<int MODE>
__global__ __launch_bounds__(256)
void gemm_bf16_nt(const uint16_t* __restrict__ A,
                  const uint16_t* __restrict__ Bw,
                  void* __restrict__ Cout) {
  __shared__ uint16_t lA[2][128 * 32];
  __shared__ uint16_t lB[2][128 * 32];

  const int tid  = threadIdx.x;
  const int lane = tid & 31;
  const int wv   = tid >> 5;
  const int hlf  = lane >> 4;
  const int l16  = lane & 15;
  const int mw   = (wv & 3) << 5;   // 0,32,64,96
  const int nw   = (wv >> 2) << 6;  // 0,64
  const long mb  = (long)blockIdx.y * 128;
  const long nb  = (long)blockIdx.x * 128;

  // staging: 128 rows x 2 segs of 16 halves for both A and B tiles
  const int sr = tid >> 1, sc = (tid & 1) << 4;

  f32x8 acc[2][4];
#pragma unroll
  for (int i = 0; i < 2; ++i)
#pragma unroll
    for (int j = 0; j < 4; ++j) acc[i][j] = zf8();

  const uint16_t* gA = A  + (mb + sr) * (long)En + sc;
  const uint16_t* gB = Bw + (nb + sr) * (long)En + sc;

  auto stage = [&](int buf, int kb) {
    g2l_16B(gA + kb,     &lA[buf][sr * 32 + sc]);
    g2l_16B(gA + kb + 8, &lA[buf][sr * 32 + sc + 8]);
    g2l_16B(gB + kb,     &lB[buf][sr * 32 + sc]);
    g2l_16B(gB + kb + 8, &lB[buf][sr * 32 + sc + 8]);
  };

  stage(0, 0);
  int buf = 0;
  for (int kb = 0; kb < En; kb += 32, buf ^= 1) {
    wait_g2l();
    __syncthreads();
    if (kb + 32 < En) stage(buf ^ 1, kb + 32);

    Frag a[2], b[4];
#pragma unroll
    for (int i = 0; i < 2; ++i) {
      const int row = mw + i * 16 + l16;
      a[i].h[0] = *(const bf16x8*)&lA[buf][row * 32 + 8 * hlf];
      a[i].h[1] = *(const bf16x8*)&lA[buf][row * 32 + 16 + 8 * hlf];
    }
#pragma unroll
    for (int j = 0; j < 4; ++j) {
      const int col = nw + j * 16 + l16;
      b[j].h[0] = *(const bf16x8*)&lB[buf][col * 32 + 8 * hlf];
      b[j].h[1] = *(const bf16x8*)&lB[buf][col * 32 + 16 + 8 * hlf];
    }
#pragma unroll
    for (int i = 0; i < 2; ++i)
#pragma unroll
      for (int j = 0; j < 4; ++j)
        acc[i][j] = __builtin_amdgcn_wmma_f32_16x16x32_bf16(
            false, a[i].v, false, b[j].v, (short)0, acc[i][j], false, false);
  }

  // epilogue: C-layout -> memory. Lane holds col = l16, rows v + 8*hlf.
#pragma unroll
  for (int i = 0; i < 2; ++i) {
    const long r0 = mb + mw + i * 16 + 8 * hlf;  // row of element v=0
#pragma unroll
    for (int j = 0; j < 4; ++j) {
      const long col = nb + nw + j * 16 + l16;
      if (MODE == 2) {
        float* o = (float*)Cout;
#pragma unroll
        for (int v = 0; v < 8; ++v) o[(r0 + v) * En + col] = acc[i][j][v];
      } else if (MODE == 0) {
        bf16_t* o = (bf16_t*)Cout;
#pragma unroll
        for (int v = 0; v < 8; ++v) o[(r0 + v) * En + col] = (bf16_t)acc[i][j][v];
      } else {
        // Vt[b][col][s]; 8 consecutive s-values per lane -> one 16B store
        bf16_t* o = (bf16_t*)Cout;
        const long bi = r0 >> 10;        // Sn == 1024
        const long s0 = r0 & (Sn - 1);
        bf16x8 pk;
#pragma unroll
        for (int v = 0; v < 8; ++v) pk[v] = (bf16_t)acc[i][j][v];
        *(bf16x8*)(o + (bi * En + col) * Sn + s0) = pk;
      }
    }
  }
}

// ---------------------------------------------------------------------------
// Flash attention: each wave owns 16 query rows of one (b,h); online softmax.
// Q,K bf16 [Mn][En] (head = 80-col slice), Vt bf16 [B][En][Sn], mask f32.
// 64-key steps: 12 WMMA (QK^T, K=80 padded to 96) + 10 WMMA (P·V, K=64).
// Row stats via DPP reductions (no LDS shuffles).
// ---------------------------------------------------------------------------
__global__ __launch_bounds__(256)
void attn_fwd(const uint16_t* __restrict__ Q,
              const uint16_t* __restrict__ Kt,
              const uint16_t* __restrict__ Vt,
              const float*    __restrict__ Msk,
              uint16_t* __restrict__ Ctx) {
  __shared__ uint16_t lP[8][16 * 64];  // per-wave prob transpose scratch

  const int tid  = threadIdx.x;
  const int lane = tid & 31;
  const int wv   = tid >> 5;
  const int hlf  = lane >> 4;
  const int l16  = lane & 15;
  const int b    = blockIdx.y >> 4;
  const int h    = blockIdx.y & 15;
  const int q0   = blockIdx.x * 128 + wv * 16;

  const float scale = 0.1118033988749895f;  // 1/sqrt(80)
  const float LOG2E = 1.4426950408889634f;

  // Q A-fragments: 16 rows x 80 cols, zero-padded to K=96 (3 chunks of 32)
  Frag aq[3];
  {
    const uint16_t* qp = Q + ((long)(b * Sn + q0 + l16)) * En + h * Dn;
#pragma unroll
    for (int c = 0; c < 3; ++c) {
      aq[c].h[0] = *(const bf16x8*)(qp + c * 32 + 8 * hlf);
      if (c < 2) {
        aq[c].h[1] = *(const bf16x8*)(qp + c * 32 + 16 + 8 * hlf);
      } else {
#pragma unroll
        for (int w = 4; w < 8; ++w) aq[c].w[w] = 0u;
      }
    }
  }

  float mst[8], lst[8];
#pragma unroll
  for (int v = 0; v < 8; ++v) { mst[v] = -__builtin_inff(); lst[v] = 0.f; }
  f32x8 acc[5];
#pragma unroll
  for (int d = 0; d < 5; ++d) acc[d] = zf8();

  const uint16_t* kbase = Kt + ((long)(b * Sn)) * En + h * Dn;
  const uint16_t* vbase = Vt + ((long)(b * En + h * Dn)) * Sn;
  const float*    mbase = Msk + ((long)(b * Sn) + q0 + 8 * hlf) * Sn;

  bf16_t* pbuf = (bf16_t*)&lP[wv][0];

  for (int kb = 0; kb < Sn; kb += 64) {
    if (kb + 64 < Sn) {  // prefetch next K/V block into cache
      __builtin_prefetch(kbase + (long)(kb + 64 + l16) * En, 0, 0);
      __builtin_prefetch(vbase + (long)l16 * Sn + kb + 64, 0, 0);
    }

    // ---- scores: four 16x16 tiles over keys [kb, kb+64) ----
    f32x8 sf[4];
#pragma unroll
    for (int t = 0; t < 4; ++t) {
      f32x8 s = zf8();
      const uint16_t* kp = kbase + ((long)(kb + t * 16 + l16)) * En;
#pragma unroll
      for (int c = 0; c < 3; ++c) {
        Frag bk;
        bk.h[0] = *(const bf16x8*)(kp + c * 32 + 8 * hlf);
        if (c == 2) {
#pragma unroll
          for (int w = 4; w < 8; ++w) bk.w[w] = 0u;  // k >= 80 pad
        } else {
          bk.h[1] = *(const bf16x8*)(kp + c * 32 + 16 + 8 * hlf);
        }
        s = __builtin_amdgcn_wmma_f32_16x16x32_bf16(
            false, aq[c].v, false, bk.v, (short)0, s, false, false);
      }
      const float* mp = mbase + kb + t * 16 + l16;
#pragma unroll
      for (int v = 0; v < 8; ++v) s[v] = s[v] * scale + mp[(long)v * Sn];
      sf[t] = s;
    }

    // ---- online softmax (row lives in one 16-lane half; DPP reduce) ----
    float alv[8];
#pragma unroll
    for (int v = 0; v < 8; ++v) {
      float mx = fmaxf(fmaxf(sf[0][v], sf[1][v]), fmaxf(sf[2][v], sf[3][v]));
      mx = rowmax16(mx);
      const float mnew = fmaxf(mst[v], mx);
      const float al = __builtin_amdgcn_exp2f((mst[v] - mnew) * LOG2E);
      mst[v] = mnew;
      float rs = 0.f;
#pragma unroll
      for (int t = 0; t < 4; ++t) {
        const float pt = __builtin_amdgcn_exp2f((sf[t][v] - mnew) * LOG2E);
        sf[t][v] = pt;
        rs += pt;
      }
      rs = rowsum16(rs);
      lst[v] = lst[v] * al + rs;
      alv[v] = al;
    }
#pragma unroll
    for (int d = 0; d < 5; ++d)
#pragma unroll
      for (int v = 0; v < 8; ++v) acc[d][v] *= alv[v];

    // ---- P: f32 C-layout -> bf16 A-layout via per-wave LDS transpose ----
#pragma unroll
    for (int t = 0; t < 4; ++t)
#pragma unroll
      for (int v = 0; v < 8; ++v)
        pbuf[(v + 8 * hlf) * 64 + t * 16 + l16] = (bf16_t)sf[t][v];
    asm volatile("s_wait_dscnt 0" ::: "memory");

    Frag pa[2];
#pragma unroll
    for (int cc = 0; cc < 2; ++cc) {
      pa[cc].h[0] = *(const bf16x8*)(pbuf + l16 * 64 + cc * 32 + 8 * hlf);
      pa[cc].h[1] = *(const bf16x8*)(pbuf + l16 * 64 + cc * 32 + 16 + 8 * hlf);
    }

    // ---- acc += P @ V  (V transposed: contiguous 8-half runs) ----
#pragma unroll
    for (int d = 0; d < 5; ++d) {
#pragma unroll
      for (int cc = 0; cc < 2; ++cc) {
        const uint16_t* vp =
            vbase + ((long)(d * 16 + l16)) * Sn + kb + cc * 32 + 8 * hlf;
        Frag bv;
        bv.h[0] = *(const bf16x8*)(vp);
        bv.h[1] = *(const bf16x8*)(vp + 16);
        acc[d] = __builtin_amdgcn_wmma_f32_16x16x32_bf16(
            false, pa[cc].v, false, bv.v, (short)0, acc[d], false, false);
      }
    }
  }

  // ---- epilogue: normalize and store context bf16 [Mn][En] ----
  bf16_t* op = (bf16_t*)Ctx;
#pragma unroll
  for (int d = 0; d < 5; ++d)
#pragma unroll
    for (int v = 0; v < 8; ++v) {
      const long row = (long)b * Sn + q0 + 8 * hlf + v;
      op[row * En + h * Dn + d * 16 + l16] = (bf16_t)(acc[d][v] / lst[v]);
    }
}

// ---------------------------------------------------------------------------
extern "C" void kernel_launch(void* const* d_in, const int* in_sizes, int n_in,
                              void* d_out, int out_size, void* d_ws, size_t ws_size,
                              hipStream_t stream) {
  (void)in_sizes; (void)n_in; (void)out_size; (void)ws_size;
  const float* x  = (const float*)d_in[0];
  const float* mk = (const float*)d_in[1];
  const float* wq = (const float*)d_in[2];
  const float* wk = (const float*)d_in[3];
  const float* wv = (const float*)d_in[4];
  const float* wo = (const float*)d_in[5];

  // workspace carve-up (bf16 halves), ~118 MB total
  uint16_t* p   = (uint16_t*)d_ws;
  uint16_t* Xh  = p; p += (size_t)Mn * En;
  uint16_t* Wqh = p; p += (size_t)En * En;
  uint16_t* Wkh = p; p += (size_t)En * En;
  uint16_t* Wvh = p; p += (size_t)En * En;
  uint16_t* Woh = p; p += (size_t)En * En;
  uint16_t* Qh  = p; p += (size_t)Mn * En;
  uint16_t* Kh  = p; p += (size_t)Mn * En;
  uint16_t* Vth = p; p += (size_t)Mn * En;   // [B][E][S]
  uint16_t* Ctx = p; p += (size_t)Mn * En;

  {
    int n4 = Mn * En / 4;
    cvt_f32_to_bf16<<<(n4 + 255) / 256, 256, 0, stream>>>(x, Xh, n4);
    n4 = En * En / 4;
    int blk = (n4 + 255) / 256;
    cvt_f32_to_bf16<<<blk, 256, 0, stream>>>(wq, Wqh, n4);
    cvt_f32_to_bf16<<<blk, 256, 0, stream>>>(wk, Wkh, n4);
    cvt_f32_to_bf16<<<blk, 256, 0, stream>>>(wv, Wvh, n4);
    cvt_f32_to_bf16<<<blk, 256, 0, stream>>>(wo, Woh, n4);
  }

  dim3 gg(En / 128, Mn / 128);  // 10 x 64 workgroups, 256 threads (8 waves)
  gemm_bf16_nt<0><<<gg, 256, 0, stream>>>(Xh, Wqh, Qh);
  gemm_bf16_nt<0><<<gg, 256, 0, stream>>>(Xh, Wkh, Kh);
  gemm_bf16_nt<1><<<gg, 256, 0, stream>>>(Xh, Wvh, Vth);

  dim3 ga(Sn / 128, Bn * Hn);  // 8 x 128
  attn_fwd<<<ga, 256, 0, stream>>>(Qh, Kh, Vth, mk, Ctx);

  gemm_bf16_nt<2><<<gg, 256, 0, stream>>>(Ctx, Woh, d_out);
}
